// EDFFN_9363028705905
// MI455X (gfx1250) — compile-verified, matrix operand order checked by reading, not run on hardware
//
#include <hip/hip_runtime.h>
#include <cstddef>
#include <cstdint>

// ---------------- types ----------------
typedef __attribute__((ext_vector_type(8)))  float  v8f;
typedef __attribute__((ext_vector_type(8)))  __bf16 v8bf;
typedef __attribute__((ext_vector_type(16))) __bf16 v16bf;
typedef int v4i __attribute__((ext_vector_type(4)));

// ---------------- problem constants ----------------
#define IMG   256
#define DIM   128          // input/output channels
#define HID   384          // hidden
#define TS    16           // spatial tile (contains four 8x8 FFT patches)
#define HALO  18           // TS + 2 (3x3 dwconv halo)
#define NPIX  (HALO*HALO)  // 324
#define XSTR  136          // sX / sA row stride in bf16 elems (128 + 8 pad) -> 272B rows, 16B aligned
#define HSTR  337          // sH row stride in floats (337 % 64 = 17 -> conflict-free channel walks)
#define YSTR  257          // sY row stride in floats

// K1 dynamic-LDS layout (bytes)
constexpr int LDS1_X  = 0;                      // bf16 [336][XSTR]  (rows 324..335 zero pad for N-tiling to 336)
constexpr int LDS1_A  = 336 * XSTR * 2;         //  91392: bf16 [64][XSTR]  W_in chunk (32 x1-rows + 32 x2-rows)
constexpr int LDS1_DW = LDS1_A + 64 * XSTR * 2; // +17408: float [64][9]    dw weights chunk
constexpr int LDS1_H  = LDS1_DW + 64 * 9 * 4;   // + 2304: float [64][HSTR] h chunk over halo
constexpr int LDS1_SZ = LDS1_H + 64 * HSTR * 4; // total 197376 B  (< 320KB WGP LDS)

// K2 dynamic-LDS layout (bytes): double-buffered G/W for async pipelining
constexpr int GBUF    = 256 * 72 * 2;           // 36864 per buffer
constexpr int WBUF    = 128 * 72 * 2;           // 18432 per buffer
constexpr int LDS2_G  = 0;                      // bf16 [2][256][72]
constexpr int LDS2_W  = 2 * GBUF;               //  73728: bf16 [2][128][72]
constexpr int LDS2_F  = LDS2_W + 2 * WBUF;      // 110592: float [128][40] fft filter
constexpr int LDS2_Y  = LDS2_F + DIM * 40 * 4;  // 131072: float [128][YSTR] y tile
constexpr int LDS2_SZ = LDS2_Y + DIM * YSTR * 4;// total 262656 B

// ---------------- CDNA5 async global->LDS copy (ASYNCcnt path) ----------------
#if defined(__AMDGCN__) && __has_builtin(__builtin_amdgcn_global_load_async_to_lds_b128)
#define CDNA5_ASYNC 1
#else
#define CDNA5_ASYNC 0
#endif

__device__ __forceinline__ void cp16_g2l(void* lds_dst, const void* gsrc) {
#if CDNA5_ASYNC
  // param pointee is int __vector(4); clang handles the generic->global/LDS addrspace casts
  __builtin_amdgcn_global_load_async_to_lds_b128((v4i*)(void*)gsrc, (v4i*)lds_dst, 0, 0);
#else
  *(uint4*)lds_dst = *(const uint4*)gsrc;   // fallback through VGPRs
#endif
}

__device__ __forceinline__ void wait_async() {
#if defined(__AMDGCN__) && __has_builtin(__builtin_amdgcn_s_wait_asynccnt)
  __builtin_amdgcn_s_wait_asynccnt(0);
#elif CDNA5_ASYNC
  asm volatile("s_wait_asynccnt 0x0" ::: "memory");
#endif
}

// ---------------- WMMA fragment gather ----------------
// 16-bit A/B layout (ISA 7.12.2): lane<16 -> K in {k..k+7} U {k+16..k+23}; lane>=16 -> +8.
// Rows are 16B-aligned (XSTR*2=272, 72*2=144 both multiples of 16).
__device__ __forceinline__ v16bf load_frag(const __bf16* p) {
  v8bf lo = *(const v8bf*)(p);
  v8bf hi = *(const v8bf*)(p + 16);
  return __builtin_shufflevector(lo, hi, 0,1,2,3,4,5,6,7,8,9,10,11,12,13,14,15);
}

// ---------------- 8-point complex FFT (DIF, radix-2), SGN=-1 fwd / +1 inv (unnormalized) ----------
struct cpx { float r, i; };

template<int SGN>
__device__ __forceinline__ void fft8(cpx x[8]) {
  const float C = 0.70710678118654752440f;
  const float S = (float)SGN;
  cpx t;
  // stage 1 (span 4), twiddles W8^k = (cos, S*sin)
  t.r = x[0].r - x[4].r; t.i = x[0].i - x[4].i; x[0].r += x[4].r; x[0].i += x[4].i; x[4] = t;
  t.r = x[1].r - x[5].r; t.i = x[1].i - x[5].i; x[1].r += x[5].r; x[1].i += x[5].i;
  x[5].r =  C*t.r - S*C*t.i;  x[5].i = S*C*t.r + C*t.i;
  t.r = x[2].r - x[6].r; t.i = x[2].i - x[6].i; x[2].r += x[6].r; x[2].i += x[6].i;
  x[6].r = -S*t.i;            x[6].i = S*t.r;
  t.r = x[3].r - x[7].r; t.i = x[3].i - x[7].i; x[3].r += x[7].r; x[3].i += x[7].i;
  x[7].r = -C*t.r - S*C*t.i;  x[7].i = S*C*t.r - C*t.i;
  // stage 2 (span 2), twiddle W4^1 = (0, S)
  t.r = x[0].r - x[2].r; t.i = x[0].i - x[2].i; x[0].r += x[2].r; x[0].i += x[2].i; x[2] = t;
  t.r = x[1].r - x[3].r; t.i = x[1].i - x[3].i; x[1].r += x[3].r; x[1].i += x[3].i;
  x[3].r = -S*t.i; x[3].i = S*t.r;
  t.r = x[4].r - x[6].r; t.i = x[4].i - x[6].i; x[4].r += x[6].r; x[4].i += x[6].i; x[6] = t;
  t.r = x[5].r - x[7].r; t.i = x[5].i - x[7].i; x[5].r += x[7].r; x[5].i += x[7].i;
  x[7].r = -S*t.i; x[7].i = S*t.r;
  // stage 3 (span 1)
  t.r = x[0].r - x[1].r; t.i = x[0].i - x[1].i; x[0].r += x[1].r; x[0].i += x[1].i; x[1] = t;
  t.r = x[2].r - x[3].r; t.i = x[2].i - x[3].i; x[2].r += x[3].r; x[2].i += x[3].i; x[3] = t;
  t.r = x[4].r - x[5].r; t.i = x[4].i - x[5].i; x[4].r += x[5].r; x[4].i += x[5].i; x[5] = t;
  t.r = x[6].r - x[7].r; t.i = x[6].i - x[7].i; x[6].r += x[7].r; x[6].i += x[7].i; x[7] = t;
  // bit-reversal: swap (1,4), (3,6)
  t = x[1]; x[1] = x[4]; x[4] = t;
  t = x[3]; x[3] = x[6]; x[6] = t;
}

// ======================================================================
// K1: fused project_in (bf16 WMMA) + 3x3 depthwise + exact-GELU gate
//     writes g [B][pix][HID] bf16 to workspace
// ======================================================================
__global__ void __launch_bounds__(256) edffn_k1(
    const float* __restrict__ x, const float* __restrict__ w_in,
    const float* __restrict__ w_dw, __bf16* __restrict__ g_ws)
{
  extern __shared__ char smem[];
  __bf16* sX  = (__bf16*)(smem + LDS1_X);
  __bf16* sA  = (__bf16*)(smem + LDS1_A);
  float*  sDW = (float*) (smem + LDS1_DW);
  float*  sH  = (float*) (smem + LDS1_H);

  const int tid  = threadIdx.x;
  const int wv   = tid >> 5;          // 8 waves (wave32)
  const int lane = tid & 31;
  const int lh   = lane >> 4;
  const int ln   = lane & 15;
  const int tx = blockIdx.x, ty = blockIdx.y, b = blockIdx.z;
  const int gx0 = tx * TS - 1, gy0 = ty * TS - 1;

  // ---- stage x halo tile (pixel-major bf16; SAME conv => zero outside image) ----
  for (int idx = tid; idx < NPIX * DIM; idx += 256) {
    int c  = idx / NPIX;
    int p  = idx - c * NPIX;
    int py = p / HALO, px = p - py * HALO;
    int gy = gy0 + py, gxx = gx0 + px;
    float v = 0.f;
    if ((unsigned)gy < (unsigned)IMG && (unsigned)gxx < (unsigned)IMG)
      v = x[(((size_t)b * DIM + c) * IMG + gy) * IMG + gxx];
    sX[p * XSTR + c] = (__bf16)v;
  }
  for (int idx = tid; idx < 12 * XSTR; idx += 256)    // zero pad rows 324..335 (N tiled to 336)
    sX[NPIX * XSTR + idx] = (__bf16)0.f;
  __syncthreads();

  for (int cc = 0; cc < 12; ++cc) {                   // 12 chunks of 32 gated channels
    // ---- load W_in chunk: rows [cc*32,+32) (x1) and [384+cc*32,+32) (x2) ----
    for (int idx = tid; idx < 64 * DIM; idx += 256) {
      int r = idx >> 7, k = idx & 127;
      int row = (r < 32) ? (cc * 32 + r) : (HID + cc * 32 + r - 32);
      sA[r * XSTR + k] = (__bf16)w_in[row * DIM + k];
    }
    for (int idx = tid; idx < 64 * 9; idx += 256) {
      int r = idx / 9, k = idx - r * 9;
      int row = (r < 32) ? (cc * 32 + r) : (HID + cc * 32 + r - 32);
      sDW[idx] = w_dw[row * 9 + k];
    }
    if (cc < 11) __builtin_prefetch(w_in + (size_t)(cc + 1) * 32 * DIM, 0, 1);
    __syncthreads();

    // ---- GEMM1: h[64][336] = W_chunk[64][128] @ xT[128][336]
    //      21 N-tiles over 8 waves; B fragment loaded once, shared by 4 M-tiles ----
    for (int nt = wv; nt < 21; nt += 8) {
      const int n0 = nt * 16;
      v8f acc[4];
      #pragma unroll
      for (int m = 0; m < 4; ++m)
        #pragma unroll
        for (int j = 0; j < 8; ++j) acc[m][j] = 0.f;
      #pragma unroll
      for (int kk = 0; kk < DIM; kk += 32) {
        v16bf bv = load_frag(sX + (n0 + ln) * XSTR + kk + lh * 8);
        #pragma unroll
        for (int m = 0; m < 4; ++m) {
          v16bf av = load_frag(sA + (m * 16 + ln) * XSTR + kk + lh * 8);
          acc[m] = __builtin_amdgcn_wmma_f32_16x16x32_bf16(false, av, false, bv,
                                                           (short)0, acc[m], false, false);
        }
      }
      #pragma unroll
      for (int m = 0; m < 4; ++m)
        #pragma unroll
        for (int j = 0; j < 8; ++j)        // C layout: VGPR j -> M = 16m+j+8*lh, N = n0+ln
          sH[(m * 16 + j + lh * 8) * HSTR + n0 + ln] = acc[m][j];
    }
    __syncthreads();

    // ---- depthwise 3x3 + exact GELU gate -> g chunk (bf16, K-contiguous pixel-major) ----
    {
      const int ch = tid & 31;
      const int ps = tid >> 5;
      const float* wk1 = sDW + ch * 9;
      const float* wk2 = sDW + (ch + 32) * 9;
      for (int q = 0; q < 32; ++q) {
        int p  = q * 8 + ps;
        int py = p >> 4, px = p & 15;
        float h1 = 0.f, h2 = 0.f;
        #pragma unroll
        for (int dy = 0; dy < 3; ++dy)
          #pragma unroll
          for (int dx = 0; dx < 3; ++dx) {
            int hp = (py + dy) * HALO + (px + dx);
            h1 += wk1[dy * 3 + dx] * sH[ ch       * HSTR + hp];
            h2 += wk2[dy * 3 + dx] * sH[(ch + 32) * HSTR + hp];
          }
        float g = 0.5f * h1 * (1.f + erff(h1 * 0.70710678118654752f)) * h2;
        size_t gp = (size_t)b * (IMG * IMG) + (size_t)(ty * TS + py) * IMG + (tx * TS + px);
        g_ws[gp * HID + cc * 32 + ch] = (__bf16)g;
      }
    }
    __syncthreads();
  }
}

// ======================================================================
// K2: project_out (bf16 WMMA, async double-buffered staging)
//     + fused per-8x8-patch spectral filter, writes final fp32 output
// ======================================================================
__global__ void __launch_bounds__(256) edffn_k2(
    const __bf16* __restrict__ g_ws, const float* __restrict__ w_out,
    const float* __restrict__ filt, float* __restrict__ out)
{
  extern __shared__ char smem[];
  float* sF = (float*)(smem + LDS2_F);
  float* sY = (float*)(smem + LDS2_Y);

  const int tid = threadIdx.x;
  const int wv = tid >> 5, lane = tid & 31, lh = lane >> 4, ln = lane & 15;
  const int tx = blockIdx.x, ty = blockIdx.y, b = blockIdx.z;
  const size_t pixbase = (size_t)b * (IMG * IMG) + (size_t)(ty * TS) * IMG + (tx * TS);

  // ---- stage helpers (lambdas keep indexing local) ----
  auto stage_g = [&](int cc, int buf) {          // raw bf16 copy: 2048 x 16B, async path
    char* dst0 = smem + LDS2_G + buf * GBUF;
    for (int idx = tid; idx < 2048; idx += 256) {
      int p = idx >> 3, seg = idx & 7;
      size_t gp = pixbase + (size_t)(p >> 4) * IMG + (p & 15);
      cp16_g2l(dst0 + p * 144 + seg * 16,
               (const char*)(g_ws + gp * HID + cc * 64) + seg * 16);
    }
  };
  auto stage_w = [&](int cc, int buf) {          // f32 -> bf16 conversion (VALU path)
    __bf16* sW = (__bf16*)(smem + LDS2_W + buf * WBUF);
    for (int idx = tid; idx < DIM * 64; idx += 256) {
      int m = idx >> 6, k = idx & 63;
      sW[m * 72 + k] = (__bf16)w_out[m * HID + cc * 64 + k];
    }
  };

  // ---- prologue: filter (async raw copy) + chunk 0 into buffer 0 ----
  for (int idx = tid; idx < 1280; idx += 256)    // 128*40 floats = 1280 x 16B
    cp16_g2l((char*)sF + idx * 16, (const char*)filt + idx * 16);
  stage_w(0, 0);
  stage_g(0, 0);
  wait_async();
  __syncthreads();

  v8f acc[16];                                   // [m-tile 0..7][n-sel 0..1]
  #pragma unroll
  for (int i = 0; i < 16; ++i)
    #pragma unroll
    for (int j = 0; j < 8; ++j) acc[i][j] = 0.f;

  for (int cc = 0; cc < 6; ++cc) {               // K = 384 in 6 chunks of 64
    const int buf = cc & 1;
    if (cc < 5) { stage_w(cc + 1, buf ^ 1); stage_g(cc + 1, buf ^ 1); }  // overlap with WMMA

    const __bf16* sG = (const __bf16*)(smem + LDS2_G + buf * GBUF);
    const __bf16* sW = (const __bf16*)(smem + LDS2_W + buf * WBUF);
    // wave wv owns N-tiles {wv, wv+8} x M-tiles 0..7; B fragments shared across 8 M-tiles
    #pragma unroll
    for (int kk = 0; kk < 64; kk += 32) {
      v16bf bv0 = load_frag(sG + ( wv      * 16 + ln) * 72 + kk + lh * 8);
      v16bf bv1 = load_frag(sG + ((wv + 8) * 16 + ln) * 72 + kk + lh * 8);
      #pragma unroll
      for (int m = 0; m < 8; ++m) {
        v16bf av = load_frag(sW + (m * 16 + ln) * 72 + kk + lh * 8);
        acc[m * 2 + 0] = __builtin_amdgcn_wmma_f32_16x16x32_bf16(false, av, false, bv0,
                                                                 (short)0, acc[m * 2 + 0], false, false);
        acc[m * 2 + 1] = __builtin_amdgcn_wmma_f32_16x16x32_bf16(false, av, false, bv1,
                                                                 (short)0, acc[m * 2 + 1], false, false);
      }
    }
    wait_async();          // next chunk's async LDS writes complete (this wave)
    __syncthreads();       // all waves' staging visible before buffer flip
  }

  // ---- spill y tile to LDS for the per-patch FFT exchange ----
  #pragma unroll
  for (int m = 0; m < 8; ++m)
    #pragma unroll
    for (int jn = 0; jn < 2; ++jn) {
      int n0 = (wv + 8 * jn) * 16;
      #pragma unroll
      for (int j = 0; j < 8; ++j)
        sY[(m * 16 + j + lh * 8) * YSTR + n0 + ln] = acc[m * 2 + jn][j];
    }
  __syncthreads();

  // ---- 4 patches x 128 channels = 512 tasks, 2 per thread ----
  for (int q = 0; q < 2; ++q) {
    int task = tid + q * 256;
    int chn  = task >> 2;
    int pt   = task & 3;
    int oy = (pt >> 1) * 8, ox = (pt & 1) * 8;
    cpx Sx[8][5];
    // forward rfft along rows (last axis)
    #pragma unroll
    for (int r = 0; r < 8; ++r) {
      cpx row[8];
      #pragma unroll
      for (int c = 0; c < 8; ++c) { row[c].r = sY[chn * YSTR + (oy + r) * 16 + ox + c]; row[c].i = 0.f; }
      fft8<-1>(row);
      #pragma unroll
      for (int v = 0; v < 5; ++v) Sx[r][v] = row[v];
    }
    // forward fft along columns, real filter multiply, inverse fft along columns
    #pragma unroll
    for (int v = 0; v < 5; ++v) {
      cpx col[8];
      #pragma unroll
      for (int u = 0; u < 8; ++u) col[u] = Sx[u][v];
      fft8<-1>(col);
      #pragma unroll
      for (int u = 0; u < 8; ++u) { float f = sF[chn * 40 + u * 5 + v]; col[u].r *= f; col[u].i *= f; }
      fft8<1>(col);
      #pragma unroll
      for (int u = 0; u < 8; ++u) Sx[u][v] = col[u];
    }
    // inverse rfft along rows (Hermitian reconstruction), scale 1/64, write out
    #pragma unroll
    for (int u = 0; u < 8; ++u) {
      cpx h[8];
      #pragma unroll
      for (int v = 0; v < 5; ++v) h[v] = Sx[u][v];
      h[5].r = Sx[u][3].r; h[5].i = -Sx[u][3].i;
      h[6].r = Sx[u][2].r; h[6].i = -Sx[u][2].i;
      h[7].r = Sx[u][1].r; h[7].i = -Sx[u][1].i;
      fft8<1>(h);
      int gy = ty * TS + oy + u;
      #pragma unroll
      for (int c = 0; c < 8; ++c)
        out[(((size_t)b * DIM + chn) * IMG + gy) * IMG + tx * TS + ox + c] = h[c].r * (1.f / 64.f);
    }
  }
}

// ======================================================================
extern "C" void kernel_launch(void* const* d_in, const int* in_sizes, int n_in,
                              void* d_out, int out_size, void* d_ws, size_t ws_size,
                              hipStream_t stream) {
  (void)in_sizes; (void)n_in; (void)out_size; (void)ws_size;
  const float* x     = (const float*)d_in[0];
  const float* w_in  = (const float*)d_in[1];
  const float* w_dw  = (const float*)d_in[2];
  const float* filt  = (const float*)d_in[3];
  const float* w_out = (const float*)d_in[4];
  float*  out  = (float*)d_out;
  __bf16* g_ws = (__bf16*)d_ws;   // 4*65536*384 bf16 = 192 MiB of scratch

  // allow large dynamic LDS (gfx1250: 320KB/WGP); idempotent, capture-safe (non-stream API)
  (void)hipFuncSetAttribute((const void*)edffn_k1, hipFuncAttributeMaxDynamicSharedMemorySize, LDS1_SZ);
  (void)hipFuncSetAttribute((const void*)edffn_k2, hipFuncAttributeMaxDynamicSharedMemorySize, LDS2_SZ);

  dim3 grid(IMG / TS, IMG / TS, 4);   // 16 x 16 x 4 = 1024 workgroups
  edffn_k1<<<grid, 256, LDS1_SZ, stream>>>(x, w_in, w_dw, g_ws);
  edffn_k2<<<grid, 256, LDS2_SZ, stream>>>(g_ws, w_out, filt, out);
}
